// TestGNN_11922829214241
// MI455X (gfx1250) — matrix operand order repaired and checked by reading, not compiled, not run on hardware
//
#include <hip/hip_runtime.h>
#include <stdint.h>

#define N_NODES 8192
#define E_EDGES 32768

typedef _Float16 v16h __attribute__((ext_vector_type(16)));
typedef float    v8f  __attribute__((ext_vector_type(8)));

// ---------------------------------------------------------------------------
// h_l = relu(edge_attr @ W1_l + b1_l), stored f16 [E,16]
// ---------------------------------------------------------------------------
__global__ __launch_bounds__(256) void edge_mlp_kernel(
    const float* __restrict__ ea, const float* __restrict__ W1,
    const float* __restrict__ b1, _Float16* __restrict__ hout)
{
    int idx = blockIdx.x * 256 + threadIdx.x;
    if (idx >= E_EDGES * 16) return;
    int e = idx >> 4, k = idx & 15;
    const float* er = ea + (size_t)e * 16;
    float acc = b1[k];
#pragma unroll
    for (int j = 0; j < 16; ++j) acc += er[j] * W1[j * 16 + k];
    hout[idx] = (_Float16)(acc > 0.f ? acc : 0.f);
}

// ---------------------------------------------------------------------------
// Repack W2 (+fold b2) into B-matrix for the fused edge GEMM, transposed f16:
//   w2p[n*KTOT + kk] = kk < 16*DIN ? W2[(kk&15)*DIN*DOUT + (kk>>4)*DOUT + n]
//                                  : b2[(kk-16*DIN)*DOUT + n]
// ---------------------------------------------------------------------------
template <int DIN, int DOUT>
__global__ __launch_bounds__(256) void w2prep_kernel(
    const float* __restrict__ W2, const float* __restrict__ b2,
    _Float16* __restrict__ w2p)
{
    constexpr int KTOT = 17 * DIN;
    int idx = blockIdx.x * 256 + threadIdx.x;
    if (idx >= DOUT * KTOT) return;
    int n = idx / KTOT, kk = idx - n * KTOT;
    float v;
    if (kk < 16 * DIN) {
        int i = kk >> 4, k = kk & 15;
        v = W2[(size_t)k * (DIN * DOUT) + i * DOUT + n];
    } else {
        v = b2[(kk - 16 * DIN) * DOUT + n];
    }
    w2p[idx] = (_Float16)v;
}

// ---------------------------------------------------------------------------
// agg[n,:] = x[n,:] @ root + bias   (accumulator init; messages atomic-add in)
// ---------------------------------------------------------------------------
template <int DIN, int DOUT>
__global__ __launch_bounds__(256) void node_root_kernel(
    const float* __restrict__ xin, const float* __restrict__ root,
    const float* __restrict__ bias, float* __restrict__ agg)
{
    int idx = blockIdx.x * 256 + threadIdx.x;
    if (idx >= N_NODES * DOUT) return;
    int n = idx / DOUT, o = idx - n * DOUT;
    const float* xr = xin + (size_t)n * DIN;
    float acc = bias[o];
#pragma unroll 8
    for (int i = 0; i < DIN; ++i) acc += xr[i] * root[i * DOUT + o];
    agg[idx] = acc;
}

// ---------------------------------------------------------------------------
// Fused NNConv message GEMM:
//   msg[e,o] = sum_kk A[e,kk] * B[kk,o],  A[e, i*16+k] = x[src_e,i]*h[e,k],
//   tail kk>=16*DIN carries b2 via A = x[src_e, kk-16*DIN].
// One wave = one (16-edge tile, 16-output tile); K swept in 32-chunks with
// v_wmma_f32_16x16x32_f16.  B (W2'^T) lives in LDS for the whole block.
// Result scattered with global_atomic_add_f32 into agg[dst].
// ---------------------------------------------------------------------------
template <int DIN, int DOUT>
__global__ __launch_bounds__(256) void msg_kernel(
    const _Float16* __restrict__ hbuf,   // [E,16] f16
    const float*    __restrict__ xin,    // [N,DIN] f32
    const int*      __restrict__ esrc,
    const int*      __restrict__ edst,
    const _Float16* __restrict__ w2p,    // [DOUT, KTOT] f16
    float*          __restrict__ agg)    // [N,DOUT] f32 (pre-initialized)
{
    constexpr int KTOT        = 17 * DIN;
    constexpr int NCHUNK      = KTOT / 32;        // 17 (din=32), 34 (din=64)
    constexpr int MAIN_CHUNKS = (16 * DIN) / 32;  // h*x part; rest = bias tail
    constexpr int NOT         = DOUT / 16;
    constexpr int PERWAVE     = 512 + 32 * DIN + 64;  // bytes: h tile + x tile + dst ids

    extern __shared__ char smem[];
    _Float16* w2t  = (_Float16*)smem;                       // DOUT*KTOT f16
    char*     wbas = smem + (size_t)DOUT * KTOT * 2;

    const int tid  = threadIdx.x;
    const int wave = tid >> 5;
    const int lane = tid & 31;
    const int g    = lane >> 4;   // half-wave group
    const int erow = lane & 15;   // edge row (A/M) == output col (B/N)

    _Float16* htile = (_Float16*)(wbas + wave * PERWAVE);        // [16][16]
    _Float16* xtile = htile + 256;                               // [16][DIN]
    int*      dtile = (int*)(wbas + wave * PERWAVE + 512 + 32 * DIN); // [16]

    // Stage shared B matrix into LDS (contiguous 16B copies).
    {
        const float4* s = (const float4*)w2p;
        float4*       d = (float4*)w2t;
        const int nv = DOUT * KTOT / 8;
        for (int i = tid; i < nv; i += 256) d[i] = s[i];
    }
    __syncthreads();

    const int ntiles = E_EDGES / 16;
    for (int tilei = blockIdx.x * 8 + wave; tilei < ntiles; tilei += gridDim.x * 8) {
        const int e0 = tilei * 16;

        // --- per-tile staging (per-wave private LDS scratch) ---
        if (lane < 16) dtile[lane] = edst[e0 + lane];
        {   // h tile: 16x16 f16, each lane copies 8 halves (16B)
            int r = lane >> 1, c = (lane & 1) * 8;
            *(float4*)(htile + r * 16 + c) =
                *(const float4*)(hbuf + (size_t)(e0 + r) * 16 + c);
        }
        {   // x tile: gather x[src_e,:] and convert f32->f16; lane covers half a row
            int s  = esrc[e0 + erow];
            const float* xr = xin + (size_t)s * DIN;
            int c0 = g * (DIN / 2);
#pragma unroll
            for (int c = 0; c < DIN / 2; ++c)
                xtile[erow * DIN + c0 + c] = (_Float16)xr[c0 + c];
        }
        {   // prefetch next tile's h rows into cache
            int nt = tilei + gridDim.x * 8;
            if (nt < ntiles)
                __builtin_prefetch(hbuf + (size_t)nt * 256 + lane * 8, 0, 3);
        }
        asm volatile("s_wait_dscnt 0" ::: "memory");  // intra-wave LDS publish

        // this lane's h row: h[erow][8g .. 8g+7] is all it ever needs
        float4 hvec = *(const float4*)(htile + erow * 16 + 8 * g);
        const _Float16* hp = (const _Float16*)&hvec;

        for (int ot = 0; ot < NOT; ++ot) {
            v8f acc = {0.f, 0.f, 0.f, 0.f, 0.f, 0.f, 0.f, 0.f};
            const _Float16* brow = w2t + (size_t)(ot * 16 + erow) * KTOT;

            for (int ch = 0; ch < NCHUNK; ++ch) {
                // ---- B fragment: two runs of 8 f16, K=8g..8g+7 and 16+8g..16+8g+7 ----
                v16h b;
                {
                    const float4* bp = (const float4*)(brow + ch * 32 + g * 8);
                    float4 b0 = bp[0], b1 = bp[2];
                    const _Float16* p0 = (const _Float16*)&b0;
                    const _Float16* p1 = (const _Float16*)&b1;
#pragma unroll
                    for (int j = 0; j < 8; ++j) { b[j] = p0[j]; b[8 + j] = p1[j]; }
                }
                // ---- A fragment ----
                v16h a;
                if (ch < MAIN_CHUNKS) {
                    // half=0: i=2ch,   K=8g..8g+7 ; half=1: i=2ch+1, same K run
#pragma unroll
                    for (int half = 0; half < 2; ++half) {
                        _Float16 xs = xtile[erow * DIN + 2 * ch + half];
#pragma unroll
                        for (int j = 0; j < 8; ++j) a[half * 8 + j] = xs * hp[j];
                    }
                } else {
                    // bias tail: A = x values directly (two contiguous runs of 8)
                    int t0 = ch * 32 - 16 * DIN + 8 * g;
                    float4 xa = *(const float4*)(xtile + erow * DIN + t0);
                    float4 xb = *(const float4*)(xtile + erow * DIN + t0 + 16);
                    const _Float16* pa = (const _Float16*)&xa;
                    const _Float16* pb = (const _Float16*)&xb;
#pragma unroll
                    for (int j = 0; j < 8; ++j) { a[j] = pa[j]; a[8 + j] = pb[j]; }
                }
                acc = __builtin_amdgcn_wmma_f32_16x16x32_f16(
                    false, a, false, b, (short)0, acc, false, false);
            }

            // ---- scatter-add the 16x16 tile: lane holds C[r+8g, erow] ----
            const int o = ot * 16 + erow;
#pragma unroll
            for (int r = 0; r < 8; ++r) {
                int d = dtile[r + 8 * g];
                atomicAdd(&agg[(size_t)d * DOUT + o], acc[r]);
            }
        }
    }
}

// ---------------------------------------------------------------------------
// x_next = LayerNorm(relu(agg)) * g + b   — one wave32 per row, shfl reduce
// ---------------------------------------------------------------------------
template <int D>
__global__ __launch_bounds__(256) void relu_ln_kernel(
    const float* __restrict__ agg, const float* __restrict__ gg,
    const float* __restrict__ bb, float* __restrict__ xout)
{
    constexpr int VPT = D / 32;
    int wave = threadIdx.x >> 5, lane = threadIdx.x & 31;
    int row = blockIdx.x * 8 + wave;
    if (row >= N_NODES) return;
    float v[VPT], s = 0.f, sq = 0.f;
#pragma unroll
    for (int p = 0; p < VPT; ++p) {
        float t = agg[(size_t)row * D + lane + p * 32];
        t = t > 0.f ? t : 0.f;
        v[p] = t; s += t; sq += t * t;
    }
#pragma unroll
    for (int m = 16; m >= 1; m >>= 1) {
        s  += __shfl_xor(s, m, 32);
        sq += __shfl_xor(sq, m, 32);
    }
    float mu  = s / (float)D;
    float var = sq / (float)D - mu * mu;
    float rst = rsqrtf(var + 1e-5f);
#pragma unroll
    for (int p = 0; p < VPT; ++p) {
        int o = lane + p * 32;
        xout[(size_t)row * D + o] = (v[p] - mu) * rst * gg[o] + bb[o];
    }
}

// ---------------------------------------------------------------------------
// post MLP + log_softmax: y = (relu(emb) @ W1 + b1) @ W2 + b2
// ---------------------------------------------------------------------------
__global__ __launch_bounds__(256) void post_kernel(
    const float* __restrict__ emb, const float* __restrict__ W1,
    const float* __restrict__ b1,  const float* __restrict__ W2,
    const float* __restrict__ b2,  float* __restrict__ out)
{
    int n = blockIdx.x * 256 + threadIdx.x;
    if (n >= N_NODES) return;
    float t[64];
#pragma unroll
    for (int o = 0; o < 64; ++o) t[o] = b1[o];
    const float* er = emb + (size_t)n * 128;
    for (int i = 0; i < 128; ++i) {
        float e = er[i]; e = e > 0.f ? e : 0.f;
        const float* wr = W1 + i * 64;
#pragma unroll
        for (int o = 0; o < 64; ++o) t[o] += e * wr[o];
    }
    float y[10];
#pragma unroll
    for (int o = 0; o < 10; ++o) y[o] = b2[o];
    for (int j = 0; j < 64; ++j) {
        float tj = t[j];
        const float* wr = W2 + j * 10;
#pragma unroll
        for (int o = 0; o < 10; ++o) y[o] += tj * wr[o];
    }
    float m = y[0];
#pragma unroll
    for (int o = 1; o < 10; ++o) m = y[o] > m ? y[o] : m;
    float se = 0.f;
#pragma unroll
    for (int o = 0; o < 10; ++o) se += __expf(y[o] - m);
    float lse = __logf(se);
#pragma unroll
    for (int o = 0; o < 10; ++o) out[(size_t)n * 10 + o] = y[o] - m - lse;
}

// ---------------------------------------------------------------------------
static constexpr size_t msg_smem(int DIN, int DOUT) {
    return (size_t)DOUT * (17 * DIN) * 2 + 8 * (size_t)(512 + 32 * DIN + 64);
}

extern "C" void kernel_launch(void* const* d_in, const int* in_sizes, int n_in,
                              void* d_out, int out_size, void* d_ws, size_t ws_size,
                              hipStream_t stream)
{
    (void)in_sizes; (void)n_in; (void)out_size; (void)ws_size;

    // --- input mapping: insertion order, params flattened per jax dict sort ---
    const float* x0    = (const float*)d_in[0];
    const int*   esrc  = (const int*)d_in[1];
    const int*   edst  = (const int*)d_in[2];
    const float* eattr = (const float*)d_in[3];
    // conv{0,1,2}: W1, W2, b1, b2, bias, root  (alphabetical: W1<W2<b1<b2<bias<root)
    const float* cW1[3]   = {(const float*)d_in[4],  (const float*)d_in[10], (const float*)d_in[16]};
    const float* cW2[3]   = {(const float*)d_in[5],  (const float*)d_in[11], (const float*)d_in[17]};
    const float* cb1[3]   = {(const float*)d_in[6],  (const float*)d_in[12], (const float*)d_in[18]};
    const float* cb2[3]   = {(const float*)d_in[7],  (const float*)d_in[13], (const float*)d_in[19]};
    const float* cbias[3] = {(const float*)d_in[8],  (const float*)d_in[14], (const float*)d_in[20]};
    const float* croot[3] = {(const float*)d_in[9],  (const float*)d_in[15], (const float*)d_in[21]};
    const float* ln_b[2]  = {(const float*)d_in[22], (const float*)d_in[24]};
    const float* ln_g[2]  = {(const float*)d_in[23], (const float*)d_in[25]};
    const float* pW1 = (const float*)d_in[26];
    const float* pW2 = (const float*)d_in[27];
    const float* pb1 = (const float*)d_in[28];
    const float* pb2 = (const float*)d_in[29];

    // --- workspace carve-out ---
    char* ws = (char*)d_ws; size_t off = 0;
    auto carve = [&](size_t bytes) -> void* {
        void* p = ws + off; off = (off + bytes + 255) & ~(size_t)255; return p;
    };
    _Float16* h[3];
    for (int l = 0; l < 3; ++l) h[l] = (_Float16*)carve((size_t)E_EDGES * 16 * 2);
    _Float16* w2p0 = (_Float16*)carve(32  * (17 * 32) * 2);
    _Float16* w2p1 = (_Float16*)carve(64  * (17 * 32) * 2);
    _Float16* w2p2 = (_Float16*)carve(128 * (17 * 64) * 2);
    float* x1   = (float*)carve((size_t)N_NODES * 32 * 4);
    float* x2   = (float*)carve((size_t)N_NODES * 64 * 4);
    float* agg0 = (float*)carve((size_t)N_NODES * 32 * 4);
    float* agg1 = (float*)carve((size_t)N_NODES * 64 * 4);

    float* emb = (float*)d_out;                 // conv2 accumulator == output emb
    float* lp  = (float*)d_out + (size_t)N_NODES * 128;

    // allow >64KB dynamic LDS for the message kernels (320KB/WGP on CDNA5)
    (void)hipFuncSetAttribute((const void*)&msg_kernel<32, 32>,
        hipFuncAttributeMaxDynamicSharedMemorySize, (int)msg_smem(32, 32));
    (void)hipFuncSetAttribute((const void*)&msg_kernel<32, 64>,
        hipFuncAttributeMaxDynamicSharedMemorySize, (int)msg_smem(32, 64));
    (void)hipFuncSetAttribute((const void*)&msg_kernel<64, 128>,
        hipFuncAttributeMaxDynamicSharedMemorySize, (int)msg_smem(64, 128));

    // --- edge MLP layer-1 for all three convs (independent of node features) ---
    for (int l = 0; l < 3; ++l)
        edge_mlp_kernel<<<(E_EDGES * 16) / 256, 256, 0, stream>>>(eattr, cW1[l], cb1[l], h[l]);

    // --- repack W2(+b2) into WMMA B layout ---
    w2prep_kernel<32, 32 ><<<(32  * 17 * 32 + 255) / 256, 256, 0, stream>>>(cW2[0], cb2[0], w2p0);
    w2prep_kernel<32, 64 ><<<(64  * 17 * 32 + 255) / 256, 256, 0, stream>>>(cW2[1], cb2[1], w2p1);
    w2prep_kernel<64, 128><<<(128 * 17 * 64 + 255) / 256, 256, 0, stream>>>(cW2[2], cb2[2], w2p2);

    // --- layer 0: 32 -> 32 ---
    node_root_kernel<32, 32><<<(N_NODES * 32) / 256, 256, 0, stream>>>(x0, croot[0], cbias[0], agg0);
    msg_kernel<32, 32><<<128, 256, msg_smem(32, 32), stream>>>(h[0], x0, esrc, edst, w2p0, agg0);
    relu_ln_kernel<32><<<N_NODES / 8, 256, 0, stream>>>(agg0, ln_g[0], ln_b[0], x1);

    // --- layer 1: 32 -> 64 ---
    node_root_kernel<32, 64><<<(N_NODES * 64) / 256, 256, 0, stream>>>(x1, croot[1], cbias[1], agg1);
    msg_kernel<32, 64><<<128, 256, msg_smem(32, 64), stream>>>(h[1], x1, esrc, edst, w2p1, agg1);
    relu_ln_kernel<64><<<N_NODES / 8, 256, 0, stream>>>(agg1, ln_g[1], ln_b[1], x2);

    // --- layer 2: 64 -> 128, accumulate straight into d_out emb region ---
    node_root_kernel<64, 128><<<(N_NODES * 128) / 256, 256, 0, stream>>>(x2, croot[2], cbias[2], emb);
    msg_kernel<64, 128><<<128, 256, msg_smem(64, 128), stream>>>(h[2], x2, esrc, edst, w2p2, emb);

    // --- post MLP + log_softmax ---
    post_kernel<<<N_NODES / 256, 256, 0, stream>>>(emb, pW1, pb1, pW2, pb2, lp);
}
